// AG_MAN_90572270338466
// MI455X (gfx1250) — matrix operand order repaired and checked by reading, not compiled, not run on hardware
//
#include <hip/hip_runtime.h>
#include <hip/hip_bf16.h>
#include <math.h>

// ---------------------------------------------------------------------------
// Problem constants (from reference)
// ---------------------------------------------------------------------------
#define BB     256
#define CIN    2048
#define COUT   512
#define HH     14
#define WW     14
#define HWN    196            // 14*14
#define MDIM   (BB * HWN)     // 50176 rows of the big GEMM
#define KTILES (CIN / 32)     // 64 k-steps of 32
#define GM     128            // block M tile
#define GN     256            // block N tile

typedef __bf16 bf16_t;
typedef bf16_t v16bf __attribute__((ext_vector_type(16)));
typedef bf16_t v8bf  __attribute__((ext_vector_type(8)));
typedef float  v8f   __attribute__((ext_vector_type(8)));
typedef int    v4i_t __attribute__((ext_vector_type(4)));

// ---------------------------------------------------------------------------
// CDNA5 async global->LDS copy (ASYNCcnt) with sync fallback
// ---------------------------------------------------------------------------
#if defined(__has_builtin)
#  if __has_builtin(__builtin_amdgcn_global_load_async_to_lds_b128)
#    define AG_HAVE_ASYNC 1
#  endif
#endif

__device__ __forceinline__ void AG_cp16(void* lds, const void* gsrc) {
#ifdef AG_HAVE_ASYNC
    __builtin_amdgcn_global_load_async_to_lds_b128(
        (v4i_t*)gsrc, (v4i_t*)lds, 0, 0);
#else
    *(v8bf*)lds = *(const v8bf*)gsrc;
#endif
}

__device__ __forceinline__ void AG_wait_async() {
#ifdef AG_HAVE_ASYNC
#  if __has_builtin(__builtin_amdgcn_s_wait_asynccnt)
    __builtin_amdgcn_s_wait_asynccnt(0);
#  else
    asm volatile("s_wait_asynccnt 0x0" ::: "memory");
#  endif
#endif
}

// ---------------------------------------------------------------------------
// Kernel 1: fold scalar w into conv1_w, convert to bf16, pack K-major per
// column: Bcol[(kt*512 + n)*32 + k] = w * conv1_w[n][kt*32+k]
// ---------------------------------------------------------------------------
__global__ void AG_prep_w(const float* __restrict__ conv1_w,
                          const float* __restrict__ wscal,
                          bf16_t* __restrict__ Bcol) {
    int idx = blockIdx.x * 256 + threadIdx.x;   // over 512*2048
    float s = wscal[0];
    int n  = idx >> 11;          // output channel
    int kg = idx & 2047;         // k
    int kt = kg >> 5, k = kg & 31;
    Bcol[((long)(kt * COUT + n)) * 32 + k] = (bf16_t)(conv1_w[idx] * s);
}

// ---------------------------------------------------------------------------
// Kernel 2: feat [b][c][hw] f32 -> A[m = b*196+hw][k = c] bf16 (LDS transpose)
// ---------------------------------------------------------------------------
__global__ void AG_feat_to_a(const float* __restrict__ feat,
                             bf16_t* __restrict__ A) {
    __shared__ bf16_t tile[HWN * 32];
    int b  = blockIdx.x >> 6;    // 64 c-tiles of 32
    int ct = blockIdx.x & 63;
    const float* src = feat + ((long)b * CIN + ct * 32) * HWN;
    for (int e = threadIdx.x; e < 32 * HWN; e += 256) {
        int c = e / HWN, hw = e % HWN;          // coalesced read along hw
        tile[hw * 32 + c] = (bf16_t)src[c * HWN + hw];
    }
    __syncthreads();
    bf16_t* dst = A + ((long)b * HWN) * CIN + ct * 32;
    for (int e = threadIdx.x; e < 32 * HWN; e += 256) {
        int hw = e >> 5, c = e & 31;            // coalesced write along c
        dst[(long)hw * CIN + c] = tile[hw * 32 + c];
    }
}

// ---------------------------------------------------------------------------
// Kernel 3: WMMA GEMM  X[m][n] = sum_k A[m][k]*Bcol[k][n] + bias[n]
// Block: 256 threads = 8 waves = 2 M-waves x 4 N-waves; block tile 128x256.
// Each wave: 64x64 tile = 4x4 accumulators, 16 v_wmma per k32 step.
// A/B tiles double-buffered in LDS, filled with async global->LDS copies.
// Fragment layouts follow CDNA5 ISA 16-bit A/B/C tables.
// ---------------------------------------------------------------------------
__global__ void __launch_bounds__(256)
AG_gemm_wmma(const bf16_t* __restrict__ A, const bf16_t* __restrict__ Bcol,
             const float* __restrict__ bias, float* __restrict__ X) {
    __shared__ __align__(16) bf16_t Ab[2][GM * 32];   // [m][k] row-major, 8KB per buf
    __shared__ __align__(16) bf16_t Bb[2][GN * 32];   // [n][k] packed,   16KB per buf

    const int tid  = threadIdx.x;
    const int lane = tid & 31;
    const int wv   = tid >> 5;
    const int mw   = wv & 1;             // 2 M-waves
    const int nw   = wv >> 1;            // 4 N-waves
    const int Mblk = blockIdx.x * GM;
    const int Nblk = blockIdx.y * GN;
    const int hf   = (lane < 16) ? 0 : 1;
    const int l15  = lane & 15;

    auto stage = [&](int kt, int buf) {
        const int k0 = kt * 32;
        // A tile: 128 rows x 32 k = 512 chunks of 16B
        #pragma unroll
        for (int i = 0; i < 2; ++i) {
            int idx = tid + i * 256;
            int m = idx >> 2, q = (idx & 3) * 8;
            AG_cp16(&Ab[buf][m * 32 + q],
                    A + (long)(Mblk + m) * CIN + k0 + q);
        }
        // B tile: 256 cols x 32 k = 1024 chunks of 16B
        #pragma unroll
        for (int i = 0; i < 4; ++i) {
            int idx = tid + i * 256;
            int n = idx >> 2, q = (idx & 3) * 8;
            AG_cp16(&Bb[buf][n * 32 + q],
                    Bcol + ((long)(kt * COUT + Nblk + n)) * 32 + q);
        }
    };

    v8f acc[4][4];
    #pragma unroll
    for (int s = 0; s < 4; ++s)
        #pragma unroll
        for (int t = 0; t < 4; ++t)
            #pragma unroll
            for (int j = 0; j < 8; ++j) acc[s][t][j] = 0.0f;

    stage(0, 0);
    for (int kt = 0; kt < KTILES; ++kt) {
        AG_wait_async();          // this wave's copies into buf[kt&1] landed
        __syncthreads();          // everyone's copies landed / prev reads done
        if (kt + 1 < KTILES) stage(kt + 1, (kt + 1) & 1);   // overlap next tile

        const bf16_t* Ac = Ab[kt & 1];
        const bf16_t* Bc = Bb[kt & 1];

        // A fragments: lanes 0-15 hold K 0..7 & 16..23, lanes 16-31 K 8..15 & 24..31
        v16bf af[4];
        #pragma unroll
        for (int s = 0; s < 4; ++s) {
            const bf16_t* p = Ac + (mw * 64 + s * 16 + l15) * 32 + hf * 8;
            v8bf lo = *(const v8bf*)p;
            v8bf hi = *(const v8bf*)(p + 16);
            #pragma unroll
            for (int j = 0; j < 8; ++j) { af[s][j] = lo[j]; af[s][j + 8] = hi[j]; }
        }
        // B fragments: lane n, lanes 0-15 K 0..7 & 16..23, lanes 16-31 K 8..15 & 24..31
        v16bf bfr[4];
        #pragma unroll
        for (int t = 0; t < 4; ++t) {
            const bf16_t* p = Bc + (nw * 64 + t * 16 + l15) * 32 + hf * 8;
            v8bf lo = *(const v8bf*)p;
            v8bf hi = *(const v8bf*)(p + 16);
            #pragma unroll
            for (int j = 0; j < 8; ++j) { bfr[t][j] = lo[j]; bfr[t][j + 8] = hi[j]; }
        }
        #pragma unroll
        for (int s = 0; s < 4; ++s)
            #pragma unroll
            for (int t = 0; t < 4; ++t)
                acc[s][t] = __builtin_amdgcn_wmma_f32_16x16x32_bf16(
                    false, af[s], false, bfr[t], (short)0, acc[s][t], false, false);
    }

    // C/D layout: lane<16 -> M=r, lane>=16 -> M=8+r; N = lane&15
    #pragma unroll
    for (int t = 0; t < 4; ++t) {
        const int n = Nblk + nw * 64 + t * 16 + l15;
        const float bv = bias[n];
        #pragma unroll
        for (int s = 0; s < 4; ++s) {
            const int mb = Mblk + mw * 64 + s * 16;
            #pragma unroll
            for (int r = 0; r < 8; ++r) {
                const int m = mb + (hf ? 8 + r : r);
                X[(long)m * COUT + n] = acc[s][t][r] + bv;
            }
        }
    }
}

// ---------------------------------------------------------------------------
// wave32 warp-dot helper
// ---------------------------------------------------------------------------
__device__ inline float AG_wredsum(float s) {
    #pragma unroll
    for (int off = 16; off; off >>= 1) s += __shfl_down(s, off, 32);
    return s;
}

// ---------------------------------------------------------------------------
// Kernel 4: a1 = swish(attr @ t1_w.T + t1_b), a2 = swish(attr @ t2_w.T + t2_b)
// ---------------------------------------------------------------------------
__global__ void AG_attr(const int* __restrict__ c, const float* __restrict__ attr_emb,
                        const float* __restrict__ t1_w, const float* __restrict__ t1_b,
                        const float* __restrict__ t2_w, const float* __restrict__ t2_b,
                        float* __restrict__ a1, float* __restrict__ a2) {
    int b = blockIdx.x;
    __shared__ float attr[512];
    int cb = c[b];
    for (int k = threadIdx.x; k < 512; k += 256) attr[k] = attr_emb[cb * 512 + k];
    __syncthreads();
    int lane = threadIdx.x & 31, wv = threadIdx.x >> 5;
    for (int n = wv; n < 512; n += 8) {
        const float* r1 = t1_w + (long)n * 512;
        const float* r2 = t2_w + (long)n * 512;
        float s1 = 0.f, s2 = 0.f;
        for (int k = lane; k < 512; k += 32) { s1 += attr[k] * r1[k]; s2 += attr[k] * r2[k]; }
        s1 = AG_wredsum(s1); s2 = AG_wredsum(s2);
        if (lane == 0) {
            float v1 = s1 + t1_b[n]; a1[b * 512 + n] = v1 / (1.f + __expf(-v1)); // swish
            float v2 = s2 + t2_b[n]; a2[b * 512 + n] = v2 / (1.f + __expf(-v2));
        }
    }
}

// ---------------------------------------------------------------------------
// Kernel 5: attmap = w1 * softmax_hw( (tanh(bn(X)) . a1) / sqrt(512) )
// ---------------------------------------------------------------------------
__global__ void AG_attmap(const float* __restrict__ X, const float* __restrict__ a1,
                          const float* __restrict__ bn_g, const float* __restrict__ bn_b,
                          const float* __restrict__ bn_m, const float* __restrict__ bn_v,
                          const float* __restrict__ w1, float* __restrict__ attw) {
    int b = blockIdx.x;
    __shared__ float aL[512], sL[512], mL[512], bL[512];
    __shared__ float raw[HWN];
    __shared__ float red[8];
    __shared__ float smax, ssum;
    for (int k = threadIdx.x; k < 512; k += 256) {
        aL[k] = a1[b * 512 + k];
        sL[k] = bn_g[k] * rsqrtf(bn_v[k] + 1e-5f);
        mL[k] = bn_m[k];
        bL[k] = bn_b[k];
    }
    __syncthreads();
    int lane = threadIdx.x & 31, wv = threadIdx.x >> 5;
    for (int hw = wv; hw < HWN; hw += 8) {
        const float* xp = X + ((long)(b * HWN + hw)) * COUT;
        float s = 0.f;
        for (int cc = lane; cc < 512; cc += 32)
            s += tanhf((xp[cc] - mL[cc]) * sL[cc] + bL[cc]) * aL[cc];
        s = AG_wredsum(s);
        if (lane == 0) raw[hw] = s * 0.04419417382415922f;   // 1/sqrt(512)
    }
    __syncthreads();
    float v = (threadIdx.x < HWN) ? raw[threadIdx.x] : -3.4e38f;
    float m0 = v;
    #pragma unroll
    for (int off = 16; off; off >>= 1) m0 = fmaxf(m0, __shfl_down(m0, off, 32));
    if (lane == 0) red[wv] = m0;
    __syncthreads();
    if (threadIdx.x == 0) { float m = red[0]; for (int i = 1; i < 8; ++i) m = fmaxf(m, red[i]); smax = m; }
    __syncthreads();
    float e = (threadIdx.x < HWN) ? __expf(v - smax) : 0.f;
    float s0 = AG_wredsum(e);
    if (lane == 0) red[wv] = s0;
    __syncthreads();
    if (threadIdx.x == 0) { float s = 0.f; for (int i = 0; i < 8; ++i) s += red[i]; ssum = s; }
    __syncthreads();
    if (threadIdx.x < HWN) attw[b * HWN + threadIdx.x] = w1[0] * e / ssum;
}

// ---------------------------------------------------------------------------
// Kernel 6: CBAM spatial attention + spatial sum
// x' = X * attw ; sa = conv7x7([avg_c x', max_c x']) ; xsum = sum_hw x'*sig(sa)
// ---------------------------------------------------------------------------
__global__ void AG_sa(const float* __restrict__ X, const float* __restrict__ attw,
                      const float* __restrict__ sa_w, float* __restrict__ xsum) {
    int b = blockIdx.x;
    __shared__ float attL[HWN], avgL[HWN], mxL[HWN], coefL[HWN], wL[98];
    int lane = threadIdx.x & 31, wv = threadIdx.x >> 5;
    if (threadIdx.x < 98)  wL[threadIdx.x]   = sa_w[threadIdx.x];
    if (threadIdx.x < HWN) attL[threadIdx.x] = attw[b * HWN + threadIdx.x];
    __syncthreads();
    for (int hw = wv; hw < HWN; hw += 8) {
        const float* xp = X + ((long)(b * HWN + hw)) * COUT;
        float s = 0.f, mn = 3.4e38f, mx = -3.4e38f;
        for (int cc = lane; cc < 512; cc += 32) {
            float x = xp[cc];
            s += x; mn = fminf(mn, x); mx = fmaxf(mx, x);
        }
        #pragma unroll
        for (int off = 16; off; off >>= 1) {
            s  += __shfl_down(s, off, 32);
            mn  = fminf(mn, __shfl_down(mn, off, 32));
            mx  = fmaxf(mx, __shfl_down(mx, off, 32));
        }
        if (lane == 0) {
            float att = attL[hw];
            avgL[hw] = att * s * (1.0f / 512.0f);
            mxL[hw]  = (att >= 0.f) ? att * mx : att * mn;   // max of scaled values
        }
    }
    __syncthreads();
    if (threadIdx.x < HWN) {
        int y = threadIdx.x / WW, x = threadIdx.x % WW;
        float s = 0.f;
        for (int dy = 0; dy < 7; ++dy) {
            int yy = y + dy - 3; if (yy < 0 || yy >= HH) continue;
            for (int dx = 0; dx < 7; ++dx) {
                int xx = x + dx - 3; if (xx < 0 || xx >= WW) continue;
                int p = yy * WW + xx;
                s += wL[dy * 7 + dx] * avgL[p] + wL[49 + dy * 7 + dx] * mxL[p];
            }
        }
        coefL[threadIdx.x] = attL[threadIdx.x] / (1.f + __expf(-s));
    }
    __syncthreads();
    for (int n = threadIdx.x; n < 512; n += 256) {
        const float* xp = X + ((long)b * HWN) * COUT + n;
        float s = 0.f;
        for (int hw = 0; hw < HWN; ++hw) s += xp[(long)hw * COUT] * coefL[hw];
        xsum[b * 512 + n] = s;
    }
}

// ---------------------------------------------------------------------------
// Kernel 7: ACA channel mask + class mask + l2 norm
// ---------------------------------------------------------------------------
__global__ void AG_final(const float* __restrict__ xsum, const float* __restrict__ a2,
                         const float* __restrict__ fc1_w, const float* __restrict__ fc1_b,
                         const float* __restrict__ fc2_w, const float* __restrict__ fc2_b,
                         const float* __restrict__ w2, const float* __restrict__ masks_w,
                         const int* __restrict__ c, float* __restrict__ out) {
    int b = blockIdx.x;
    __shared__ float v0[1024], m1[512], y[512], red[8];
    __shared__ float snorm;
    for (int k = threadIdx.x; k < 512; k += 256) {
        v0[k]       = xsum[b * 512 + k];
        v0[512 + k] = a2[b * 512 + k];
    }
    __syncthreads();
    int lane = threadIdx.x & 31, wv = threadIdx.x >> 5;
    for (int n = wv; n < 512; n += 8) {
        const float* row = fc1_w + (long)n * 1024;
        float s = 0.f;
        for (int k = lane; k < 1024; k += 32) s += v0[k] * row[k];
        s = AG_wredsum(s);
        if (lane == 0) m1[n] = fmaxf(s + fc1_b[n], 0.f);
    }
    __syncthreads();
    int cb = c[b];
    float w2v = w2[0];
    for (int n = wv; n < 512; n += 8) {
        const float* row = fc2_w + (long)n * 512;
        float s = 0.f;
        for (int k = lane; k < 512; k += 32) s += m1[k] * row[k];
        s = AG_wredsum(s);
        if (lane == 0) {
            float m2 = 1.f / (1.f + __expf(-(s + fc2_b[n])));
            y[n] = v0[n] * w2v * m2 * fmaxf(masks_w[cb * 512 + n], 0.f);
        }
    }
    __syncthreads();
    float ss = 0.f;
    for (int n = threadIdx.x; n < 512; n += 256) ss += y[n] * y[n];
    ss = AG_wredsum(ss);
    if (lane == 0) red[wv] = ss;
    __syncthreads();
    if (threadIdx.x == 0) { float s = 0.f; for (int i = 0; i < 8; ++i) s += red[i]; snorm = rsqrtf(s); }
    __syncthreads();
    for (int n = threadIdx.x; n < 512; n += 256) out[b * 512 + n] = y[n] * snorm;
}

// ---------------------------------------------------------------------------
// Launch
// ---------------------------------------------------------------------------
extern "C" void kernel_launch(void* const* d_in, const int* in_sizes, int n_in,
                              void* d_out, int out_size, void* d_ws, size_t ws_size,
                              hipStream_t stream) {
    const float* feat     = (const float*)d_in[0];
    const int*   c        = (const int*)  d_in[1];
    const float* attr_emb = (const float*)d_in[2];
    const float* t1_w     = (const float*)d_in[3];
    const float* t1_b     = (const float*)d_in[4];
    const float* conv1_w  = (const float*)d_in[5];
    const float* conv1_b  = (const float*)d_in[6];
    const float* bn_g     = (const float*)d_in[7];
    const float* bn_b     = (const float*)d_in[8];
    const float* bn_m     = (const float*)d_in[9];
    const float* bn_v     = (const float*)d_in[10];
    const float* sa_w     = (const float*)d_in[11];
    const float* t2_w     = (const float*)d_in[12];
    const float* t2_b     = (const float*)d_in[13];
    const float* fc1_w    = (const float*)d_in[14];
    const float* fc1_b    = (const float*)d_in[15];
    const float* fc2_w    = (const float*)d_in[16];
    const float* fc2_b    = (const float*)d_in[17];
    const float* w        = (const float*)d_in[18];
    const float* w1       = (const float*)d_in[19];
    const float* w2       = (const float*)d_in[20];
    const float* masks_w  = (const float*)d_in[21];
    float* out            = (float*)d_out;

    // Workspace layout
    char* ws = (char*)d_ws;
    size_t offA    = 0;                                   // A bf16: M x K
    size_t offB    = offA + (size_t)MDIM * CIN * 2;       // Bcol bf16: K x N packed
    size_t offX    = offB + (size_t)CIN * COUT * 2;       // X f32: M x N
    size_t offA1   = offX + (size_t)MDIM * COUT * 4;
    size_t offA2   = offA1 + (size_t)BB * 512 * 4;
    size_t offAtt  = offA2 + (size_t)BB * 512 * 4;
    size_t offXs   = offAtt + (size_t)BB * HWN * 4;
    bf16_t* A    = (bf16_t*)(ws + offA);
    bf16_t* Bcol = (bf16_t*)(ws + offB);
    float*  X    = (float*) (ws + offX);
    float*  a1   = (float*) (ws + offA1);
    float*  a2v  = (float*) (ws + offA2);
    float*  attw = (float*) (ws + offAtt);
    float*  xs   = (float*) (ws + offXs);

    AG_prep_w   <<<(CIN * COUT) / 256, 256, 0, stream>>>(conv1_w, w, Bcol);
    AG_feat_to_a<<<BB * (CIN / 32),    256, 0, stream>>>(feat, A);
    dim3 ggrid(MDIM / GM, COUT / GN);
    AG_gemm_wmma<<<ggrid, 256, 0, stream>>>(A, Bcol, conv1_b, X);
    AG_attr     <<<BB, 256, 0, stream>>>(c, attr_emb, t1_w, t1_b, t2_w, t2_b, a1, a2v);
    AG_attmap   <<<BB, 256, 0, stream>>>(X, a1, bn_g, bn_b, bn_m, bn_v, w1, attw);
    AG_sa       <<<BB, 256, 0, stream>>>(X, attw, sa_w, xs);
    AG_final    <<<BB, 256, 0, stream>>>(xs, a2v, fc1_w, fc1_b, fc2_w, fc2_b, w2, masks_w, c, out);
}
